// ChannelsSelectedConv2d_40312563040581
// MI455X (gfx1250) — compile-verified, matrix operand order checked by reading, not run on hardware
//
#include <hip/hip_runtime.h>

typedef __attribute__((ext_vector_type(16))) __bf16          v16bf;
typedef __attribute__((ext_vector_type(16))) unsigned short  v16u;
typedef __attribute__((ext_vector_type(8)))  unsigned short  v8u;
typedef __attribute__((ext_vector_type(8)))  float           v8f;
typedef __attribute__((ext_vector_type(4)))  float           v4f;

#define BB 32
#define CC 64
#define HH 112
#define WW 112
#define OO 64
#define KS 5
#define HO 108
#define WO 108
#define GROUPS 8
#define RPB 12              // output rows per block
#define YTILES 9            // 108 / 12
#define XTILES 7            // ceil(108 / 16)
#define KCH 7               // 7 K-chunks of 32
#define XROWS 16            // RPB + KS - 1
#define WPAD 120            // padded LDS x-width (x up to 115 read)
#define NT (RPB * XTILES)   // 84 tiles per block

// element offset (in unsigned shorts) of tap (r,s) from the (yloc, x0+m) base
#define EL(r, s) ((((r) * WPAD) + (s)) * 8)

// fp32 -> bf16 round-to-nearest-even
__device__ __forceinline__ unsigned short f2bf(float f) {
    unsigned u = __builtin_bit_cast(unsigned, f);
    unsigned r = u + 0x7FFFu + ((u >> 16) & 1u);
    return (unsigned short)(r >> 16);
}

// K-slot map: chunk kc, slot p (0..3) -> tap (r,s); false for pad slots
__device__ __forceinline__ bool slot_rs(int kc, int p, int& r, int& s) {
    if (kc < 5)  { r = kc; s = p; return true; }       // s = 0..3
    if (kc == 5) { r = p;  s = 4; return true; }       // s = 4, r = 0..3
    if (p == 0)  { r = 4;  s = 4; return true; }       // last tap
    return false;                                       // 3 pad slots (B = 0)
}

__device__ __forceinline__ v16bf pack16(v8u lo, v8u hv) {
    v16u u = __builtin_shufflevector(lo, hv, 0, 1, 2, 3, 4, 5, 6, 7,
                                     8, 9, 10, 11, 12, 13, 14, 15);
    return __builtin_bit_cast(v16bf, u);
}

// Load all 7 A fragments for tile (yloc, x0)
__device__ __forceinline__ void load_afrag(v16bf a[KCH], const unsigned short* Xs2,
                                           int yloc, int x0, int m, int hA, int hB) {
    const int baseidx = (yloc * WPAD + x0 + m) * 8;
    const unsigned short* pA = Xs2 + baseidx + hA;   // chunks 0-4 (+s parity)
    const unsigned short* pB = Xs2 + baseidx + hB;   // chunk 5   (+r parity)
    const unsigned short* pC = Xs2 + baseidx;        // chunk 6
    #pragma unroll
    for (int kc = 0; kc < 5; ++kc)
        a[kc] = pack16(*(const v8u*)(pA + EL(kc, 0)), *(const v8u*)(pA + EL(kc, 2)));
    a[5] = pack16(*(const v8u*)(pB + EL(0, 4)), *(const v8u*)(pB + EL(2, 4)));
    {
        v8u lo = *(const v8u*)(pC + EL(4, 4));       // other 3 slots: B = 0
        a[6] = pack16(lo, lo);
    }
}

// WMMA chain + vectorized store for one tile
__device__ __forceinline__ void compute_store(const v16bf a[KCH], const v16bf b[KCH],
                                              v8f cinit, float* outb,
                                              int y, int x0, int hiD, bool nlow) {
    v8f acc = __builtin_amdgcn_wmma_f32_16x16x32_bf16(
        false, a[0], false, b[0], (short)0, cinit, false, false);
    #pragma unroll
    for (int kc = 1; kc < KCH; ++kc)
        acc = __builtin_amdgcn_wmma_f32_16x16x32_bf16(
            false, a[kc], false, b[kc], (short)0, acc, false, false);

    const int xs = x0 + hiD;
    float* orow = outb + y * WO + xs;                // 16B-aligned (xs % 4 == 0)
    v4f lo4 = __builtin_shufflevector(acc, acc, 0, 1, 2, 3);
    v4f hi4 = __builtin_shufflevector(acc, acc, 4, 5, 6, 7);
    if (nlow) {
        *(v4f*)(orow) = lo4;                         // xs+3 <= 107 always
        if (xs + 7 < WO)                             // fails only (96, upper half)
            *(v4f*)(orow + 4) = hi4;
    }
}

__global__ __launch_bounds__(256, 1)
void ChannelsSelectedConv2d_kernel(const float* __restrict__ X,
                                   const float* __restrict__ Wt,
                                   const float* __restrict__ bias,
                                   const int*   __restrict__ sel,
                                   float* __restrict__ out)
{
    // X staged channels-innermost: Xs2[(row*WPAD + x)*8 + c], bf16 bits
    __shared__ __align__(16) unsigned short Xs2[XROWS * WPAD * 8];    // 30720 B
    // im2col weights, fragment-transposed: Bs2[(n*KCH + kc)*32 + j]
    __shared__ __align__(16) unsigned short Bs2[16 * KCH * 32];       //  7168 B

    const int tid = threadIdx.x;
    int bid = blockIdx.x;
    const int yt = bid % YTILES;  bid /= YTILES;
    const int g  = bid % GROUPS;  bid /= GROUPS;
    const int b  = bid;
    const int y0 = yt * RPB;

    // ---- Build B (once per block), laid out for b128 fragment loads ----
    for (int i = tid; i < 16 * KCH * 32; i += 256) {
        const int j  = i & 31;
        const int kc = (i >> 5) % KCH;
        const int n  = i / (32 * KCH);
        const int p  = j >> 3, c = j & 7;
        int r, s;
        unsigned short val = 0;
        if (n < 8 && slot_rs(kc, p, r, s)) {
            const int o  = g + 8 * n;     // output channel
            const int cg = g + 8 * c;     // global input channel
            int kk = 0;
            #pragma unroll
            for (int k = 0; k < 8; ++k)
                if (sel[o * 8 + k] == cg) kk = k;
            val = f2bf(Wt[((o * 8 + kk) * KS + r) * KS + s]);
        }
        Bs2[i] = val;
    }

    // ---- Stage X region: 8 channels packed per ds_store_b128 ----
    for (int i = tid; i < XROWS * WW; i += 256) {
        const int x   = i % WW;
        const int row = i / WW;
        v8u pack;
        #pragma unroll
        for (int c = 0; c < 8; ++c) {
            const int cg = g + 8 * c;
            pack[c] = f2bf(X[(((size_t)b * CC + cg) * HH + (y0 + row)) * WW + x]);
        }
        *(v8u*)&Xs2[(row * WPAD + x) * 8] = pack;
    }
    // zero the x-padding columns [112, 120)
    for (int i = tid; i < XROWS * (WPAD - WW); i += 256) {
        const int x   = WW + i % (WPAD - WW);
        const int row = i / (WPAD - WW);
        v8u z = (v8u)0;
        *(v8u*)&Xs2[(row * WPAD + x) * 8] = z;
    }
    __syncthreads();

    const int lane = tid & 31;
    const int wv   = tid >> 5;
    const int n    = lane & 15;                 // N column (oc) for B/D
    const int m    = lane & 15;                 // M row (pixel) for A
    const int half = (lane >> 4) & 1;           // lane half -> slot parity
    const int hiD  = half * 8;                  // D-row offset for upper lanes
    const int hA   = half * 8;                  // chunks 0-4: s+1 for upper half
    const int hB   = half * (WPAD * 8);         // chunk 5:   r+1 for upper half
    const bool nlow = (n < 8);

    // ---- B fragments via b128 loads (held across tiles) ----
    v16bf bfrag[KCH];
    #pragma unroll
    for (int kc = 0; kc < KCH; ++kc) {
        const unsigned short* pb = Bs2 + (n * KCH + kc) * 32 + 8 * half;
        bfrag[kc] = pack16(*(const v8u*)(pb), *(const v8u*)(pb + 16));
    }

    // bias folded into the first WMMA's C operand
    const float bv = bias[g + 8 * (n & 7)];
    v8f cinit;
    #pragma unroll
    for (int v = 0; v < 8; ++v) cinit[v] = bv;

    float* outb = out + ((size_t)b * OO + (g + 8 * (n & 7))) * (HO * WO);

    // ---- Tiles, wave-strided by 8, ping-pong double-buffered A ----
    int t = wv;
    if (t < NT) {
        int yloc = t / XTILES, xt = t - yloc * XTILES;   // one division, pre-loop

        v16bf a0[KCH], a1[KCH];
        load_afrag(a0, Xs2, yloc, xt * 16, m, hA, hB);

        while (true) {
            // next-tile coords: stride 8 = XTILES + 1
            int yloc1 = yloc + 1, xt1 = xt + 1;
            if (xt1 >= XTILES) { xt1 -= XTILES; ++yloc1; }
            const int t1 = t + 8;
            if (t1 < NT) load_afrag(a1, Xs2, yloc1, xt1 * 16, m, hA, hB);

            compute_store(a0, bfrag, cinit, outb, y0 + yloc, xt * 16, hiD, nlow);
            if (t1 >= NT) break;

            int yloc2 = yloc1 + 1, xt2 = xt1 + 1;
            if (xt2 >= XTILES) { xt2 -= XTILES; ++yloc2; }
            const int t2 = t1 + 8;
            if (t2 < NT) load_afrag(a0, Xs2, yloc2, xt2 * 16, m, hA, hB);

            compute_store(a1, bfrag, cinit, outb, y0 + yloc1, xt1 * 16, hiD, nlow);
            if (t2 >= NT) break;

            t = t2; yloc = yloc2; xt = xt2;
        }
    }
}

extern "C" void kernel_launch(void* const* d_in, const int* in_sizes, int n_in,
                              void* d_out, int out_size, void* d_ws, size_t ws_size,
                              hipStream_t stream) {
    const float* X    = (const float*)d_in[0];
    const float* Wt   = (const float*)d_in[1];
    const float* bias = (const float*)d_in[2];
    const int*   sel  = (const int*)d_in[3];
    float* outp = (float*)d_out;

    dim3 grid(BB * GROUPS * YTILES);   // 32 * 8 * 9 = 2304 workgroups
    ChannelsSelectedConv2d_kernel<<<grid, 256, 0, stream>>>(X, Wt, bias, sel, outp);
}
